// GSoftmax_15951508537566
// MI455X (gfx1250) — matrix operand order repaired and checked by reading, not compiled, not run on hardware
//
#include <hip/hip_runtime.h>
#include <hip/hip_bf16.h>

typedef __attribute__((ext_vector_type(16))) __bf16 v16bf;
typedef __attribute__((ext_vector_type(8)))  float  v8f;

#define C_CLASSES 256
#define ALPHA_C   0.1f
#define INV_TEMP  1.0f
#define EPS_C     1e-8f
#define SLICES    64
#define ROWS_PER_BLOCK 128
// ---- main-kernel LDS layout (bytes) ----
#define MEANS_STRIDE_B 520                         // 256 bf16 padded to 130 dwords
#define LDS_MEANS_OFF  0
#define LDS_MEANS_SZ   (256 * MEANS_STRIDE_B)      // 133120
#define X_STRIDE_DW    260
#define LDS_X_OFF      LDS_MEANS_SZ
#define LDS_X_SZ       (ROWS_PER_BLOCK * X_STRIDE_DW * 4)   // 133120
#define LDS_MNORM_OFF  (LDS_X_OFF + LDS_X_SZ)      // 266240
#define LDS_XNORM_OFF  (LDS_MNORM_OFF + 1024)      // 267264
#define LDS_TOTAL      (LDS_XNORM_OFF + ROWS_PER_BLOCK * 4) // 267776  (< 320 KB)

// ---------------------------------------------------------------------------
// Kernel 1: deterministic class-sum partials.
// Block (c, slice): scan slice rows; threads own one column each; accumulate
// in registers in increasing row order (bitwise deterministic, no atomics).
// ---------------------------------------------------------------------------
__global__ void __launch_bounds__(256, 4)
gsoftmax_stats(const float* __restrict__ logits, const int* __restrict__ labels,
               float* __restrict__ psum, float* __restrict__ pcnt,
               int rows_per_slice)
{
    const int c = blockIdx.x >> 6;       // class
    const int s = blockIdx.x & (SLICES - 1);
    const int t = threadIdx.x;           // column
    const int r0 = s * rows_per_slice;

    float sum = 0.f;
    float cnt = 0.f;
    for (int i = 0; i < rows_per_slice; ++i) {
        const int r = r0 + i;
        if (labels[r] == c) {            // wave-uniform branch -> scalar skip
            sum += logits[(size_t)r * C_CLASSES + t];
            cnt += 1.f;
        }
    }
    psum[(size_t)((c << 6) | s) * C_CLASSES + t] = sum;
    if (t == 0) pcnt[(c << 6) | s] = cnt;
}

// ---------------------------------------------------------------------------
// Kernel 2: fixed-order reduction of partials -> class means (bf16) + |m|^2.
// ---------------------------------------------------------------------------
__global__ void __launch_bounds__(256, 4)
gsoftmax_means(const float* __restrict__ psum, const float* __restrict__ pcnt,
               __bf16* __restrict__ meansg, float* __restrict__ mnormg)
{
    const int c = blockIdx.x;
    const int t = threadIdx.x;

    float sum = 0.f;
    #pragma unroll 4
    for (int s = 0; s < SLICES; ++s)
        sum += psum[(size_t)((c << 6) | s) * C_CLASSES + t];
    float cnt = 0.f;
    #pragma unroll 4
    for (int s = 0; s < SLICES; ++s)
        cnt += pcnt[(c << 6) | s];

    const float mean = sum / (cnt + EPS_C);
    meansg[c * C_CLASSES + t] = (__bf16)mean;

    // block-reduce sum of squares -> mnorm[c]
    float sq = mean * mean;
    #pragma unroll
    for (int off = 16; off >= 1; off >>= 1) sq += __shfl_xor(sq, off);
    __shared__ float red[8];
    if ((t & 31) == 0) red[t >> 5] = sq;
    __syncthreads();
    if (t < 8) {
        float v = red[t];
        #pragma unroll
        for (int off = 4; off >= 1; off >>= 1) v += __shfl_xor(v, off);
        if (t == 0) mnormg[c] = v;
    }
}

// ---------------------------------------------------------------------------
// Kernel 3: fused main pass.  out = softmax(x/T - alpha * cdist(x, means))
// P = X * means^T via v_wmma_f32_16x16x32_bf16, means resident in LDS.
// ---------------------------------------------------------------------------
__global__ void __launch_bounds__(256, 1)
gsoftmax_main(const float* __restrict__ logits,
              const __bf16* __restrict__ meansg,
              const float* __restrict__ mnormg,
              float* __restrict__ out)
{
    extern __shared__ char smem[];
    const int tid  = threadIdx.x;
    const int wave = tid >> 5;
    const int lane = tid & 31;

    // ---- phase 1: means (bf16) -> LDS, rows padded to 520 B --------------
    {
        const uint2* src = (const uint2*)meansg;   // 64 uint2 per row
        const int e  = tid & 63;
        const int r0 = tid >> 6;                   // 0..3
        #pragma unroll 4
        for (int it = 0; it < 64; ++it) {
            const int row = it * 4 + r0;
            *(uint2*)(smem + LDS_MEANS_OFF + row * MEANS_STRIDE_B + e * 8) =
                src[row * 64 + e];
        }
    }
    // ---- phase 2: mnorm -> LDS ------------------------------------------
    ((float*)(smem + LDS_MNORM_OFF))[tid] = mnormg[tid];

    // ---- phase 3: x tile (f32) -> LDS + per-row squared norms ------------
    {
        float* xw = (float*)(smem + LDS_X_OFF) + wave * 16 * X_STRIDE_DW;
        float* xn = (float*)(smem + LDS_XNORM_OFF);
        const int rowg0 = blockIdx.x * ROWS_PER_BLOCK + wave * 16;
        for (int i = 0; i < 16; ++i) {
            const float* src = logits + (size_t)(rowg0 + i) * C_CLASSES + lane * 8;
            float4 a = *(const float4*)(src);
            float4 b = *(const float4*)(src + 4);
            *(float4*)(xw + i * X_STRIDE_DW + lane * 8)     = a;
            *(float4*)(xw + i * X_STRIDE_DW + lane * 8 + 4) = b;
            float ss = a.x*a.x + a.y*a.y + a.z*a.z + a.w*a.w
                     + b.x*b.x + b.y*b.y + b.z*b.z + b.w*b.w;
            #pragma unroll
            for (int off = 16; off >= 1; off >>= 1) ss += __shfl_xor(ss, off);
            if (lane == 0) xn[wave * 16 + i] = ss;
        }
    }
    __syncthreads();

    // ---- phase 4: P[16x256] = X[16x256] * means^T via WMMA ---------------
    v8f acc[16];
    {
        const v8f zero = {0.f,0.f,0.f,0.f,0.f,0.f,0.f,0.f};
        #pragma unroll
        for (int j = 0; j < 16; ++j) acc[j] = zero;
    }
    const float* xr = (const float*)(smem + LDS_X_OFF)
                    + (wave * 16 + (lane & 15)) * X_STRIDE_DW;
    const int aSel = (lane >> 4) & 1;   // upper half-wave: A K-offset +16
    const int bOff = aSel << 4;         // upper half-wave: B K-offset +16

    #pragma unroll
    for (int s = 0; s < 8; ++s) {       // K = 256 in steps of 32
        const int k0 = s * 32 + aSel * 16;
        float4 lo0 = *(const float4*)(xr + k0);
        float4 lo1 = *(const float4*)(xr + k0 + 4);
        float4 hi0 = *(const float4*)(xr + k0 + 8);
        float4 hi1 = *(const float4*)(xr + k0 + 12);
        v16bf a;
        a[0]=(__bf16)lo0.x;  a[1]=(__bf16)lo0.y;  a[2]=(__bf16)lo0.z;  a[3]=(__bf16)lo0.w;
        a[4]=(__bf16)lo1.x;  a[5]=(__bf16)lo1.y;  a[6]=(__bf16)lo1.z;  a[7]=(__bf16)lo1.w;
        a[8]=(__bf16)hi0.x;  a[9]=(__bf16)hi0.y;  a[10]=(__bf16)hi0.z; a[11]=(__bf16)hi0.w;
        a[12]=(__bf16)hi1.x; a[13]=(__bf16)hi1.y; a[14]=(__bf16)hi1.z; a[15]=(__bf16)hi1.w;
        #pragma unroll
        for (int j = 0; j < 16; ++j) {  // 16 column tiles
            const char* bp = smem + LDS_MEANS_OFF
                           + (j * 16 + (lane & 15)) * MEANS_STRIDE_B
                           + (s * 32 + bOff) * 2;
            v16bf b;
            __builtin_memcpy(&b, bp, sizeof(b));
            acc[j] = __builtin_amdgcn_wmma_f32_16x16x32_bf16(
                false, a, false, b, (short)0, acc[j], false, false);
        }
    }

    // ---- phase 5: fused epilogue: softmax(x/T - alpha*dist) --------------
    const float* xw2 = (const float*)(smem + LDS_X_OFF) + wave * 16 * X_STRIDE_DW;
    const float* mn  = (const float*)(smem + LDS_MNORM_OFF);
    const float* xnm = (const float*)(smem + LDS_XNORM_OFF);
    const int lh = lane & 15;
    const int rb = (lane < 16) ? 0 : 8;

    for (int v = 0; v < 8; ++v) {
        const int row = rb + v;
        const float xn = xnm[wave * 16 + row];
        float sv[16];
        float mx = -3.402823466e38f;
        #pragma unroll
        for (int j = 0; j < 16; ++j) {
            const int c = j * 16 + lh;
            const float p  = acc[j][v];
            const float d2 = xn + mn[c] - 2.0f * p;
            const float d  = __builtin_sqrtf(fmaxf(d2, 0.0f));
            const float sc = xw2[row * X_STRIDE_DW + c] * INV_TEMP - ALPHA_C * d;
            sv[j] = sc;
            mx = fmaxf(mx, sc);
        }
        #pragma unroll
        for (int off = 8; off >= 1; off >>= 1) mx = fmaxf(mx, __shfl_xor(mx, off));
        float sum = 0.f;
        #pragma unroll
        for (int j = 0; j < 16; ++j) {
            const float e = __expf(sv[j] - mx);
            sv[j] = e;
            sum += e;
        }
        #pragma unroll
        for (int off = 8; off >= 1; off >>= 1) sum += __shfl_xor(sum, off);
        const float inv = 1.0f / sum;
        const size_t obase =
            (size_t)(blockIdx.x * ROWS_PER_BLOCK + wave * 16 + row) * C_CLASSES;
        #pragma unroll
        for (int j = 0; j < 16; ++j) out[obase + j * 16 + lh] = sv[j] * inv;
    }
}

// ---------------------------------------------------------------------------
extern "C" void kernel_launch(void* const* d_in, const int* in_sizes, int n_in,
                              void* d_out, int out_size, void* d_ws, size_t ws_size,
                              hipStream_t stream) {
    const float* logits = (const float*)d_in[0];
    const int*   labels = (const int*)d_in[1];
    float*       out    = (float*)d_out;

    const int n = in_sizes[0] / C_CLASSES;          // 262144
    const int rows_per_slice = n / SLICES;          // 4096

    // workspace layout
    char* ws = (char*)d_ws;
    float*  psum   = (float*)ws;                                   // 16 MB
    float*  pcnt   = (float*)(ws + (size_t)SLICES * C_CLASSES * C_CLASSES * 4);
    float*  mnormg = (float*)((char*)pcnt + C_CLASSES * SLICES * 4);
    __bf16* meansg = (__bf16*)((char*)mnormg + C_CLASSES * 4);

    gsoftmax_stats<<<C_CLASSES * SLICES, 256, 0, stream>>>(
        logits, labels, psum, pcnt, rows_per_slice);
    gsoftmax_means<<<C_CLASSES, 256, 0, stream>>>(psum, pcnt, meansg, mnormg);
    gsoftmax_main<<<n / ROWS_PER_BLOCK, 256, LDS_TOTAL, stream>>>(
        logits, meansg, mnormg, out);
}
